// Attention_65609920414302
// MI455X (gfx1250) — matrix operand order repaired and checked by reading, not compiled, not run on hardware
//
#include <hip/hip_runtime.h>

// ---------------------------------------------------------------------------
// MI455X (gfx1250) CDNA5 implementation.
//  - All heavy matmuls on v_wmma_f32_16x16x32_bf16 (wave32 WMMA).
//  - GEMMs are software-pipelined with double-buffered LDS: the Tensor Data
//    Mover (tensor_load_to_lds) streams the next A tile while WMMAs run on
//    the current tile; s_wait_tensorcnt drains after compute.
//  - global_prefetch on tile t+2 of B.
// Fragment layouts used here (contiguous k, same order in A and B):
//   A (16x32 bf16): lane l holds row M=l%16; elem e -> K = e + 16*(l/16)
//   B (32x16 bf16): lane l holds col N=l%16; elem e -> K = e + 16*(l/16)
//   C/D (16x16 f32): reg r, lane l -> M = r + 8*(l/16), N = l%16
// ---------------------------------------------------------------------------

typedef unsigned short u16;
typedef __attribute__((ext_vector_type(16))) __bf16 v16bf;
typedef __attribute__((ext_vector_type(8)))  float  v8f;

#define T_SEQ  1024
#define DMODEL 1024
#define NBR    4
#define NHEAD  16
#define HTOT   64
#define DHEAD  64
#define QDIM   4096

__device__ __forceinline__ u16 f2bf(float f) {
  unsigned u = __float_as_uint(f);
  u += 0x7FFFu + ((u >> 16) & 1u);      // round to nearest even
  return (u16)(u >> 16);
}

union Frag16 {
  v16bf  v;
  float4 f4[2];
  u16    u[16];
};

__device__ __forceinline__ v8f v8f_zero() {
  v8f z = {0.f, 0.f, 0.f, 0.f, 0.f, 0.f, 0.f, 0.f};
  return z;
}

// --------------------- Tensor Data Mover (TDM) helper ----------------------
#if defined(__has_builtin)
#if __has_builtin(__builtin_amdgcn_tensor_load_to_lds)
#define HAVE_TDM 1
#endif
#endif

#ifdef HAVE_TDM
typedef __attribute__((ext_vector_type(4))) unsigned tdm_u4;
typedef __attribute__((ext_vector_type(8))) int      tdm_i8;
typedef __attribute__((ext_vector_type(4))) int      tdm_i4;

// 2D tile (bf16): tile_k contiguous elements x tile_m rows, row stride
// stride_k elements, into LDS at byte offset lds_off (row-major [m][tile_k]).
// D# packing per CDNA5 ISA 8.3 (group0) / 8.4 (group1).
__device__ __forceinline__ void
tdm_load_2d_to_lds(unsigned lds_off, const void* gptr,
                   unsigned tile_k, unsigned tile_m,
                   unsigned tensor_k, unsigned tensor_m,
                   unsigned long long stride_k) {
  unsigned long long ga = (unsigned long long)(size_t)gptr;
  tdm_u4 g0;
  g0[0] = 1u;                                          // count=1 (user, valid)
  g0[1] = lds_off;                                     // lds_addr (bytes)
  g0[2] = (unsigned)ga;                                // global_addr[31:0]
  g0[3] = (unsigned)((ga >> 32) & 0x01FFFFFFu)         // global_addr[56:32]
        | (2u << 30);                                  // type=2 ("image")
  tdm_i8 g1;
  g1[0] = (int)(1u << 16);                             // data_size=1 (2 bytes)
  g1[1] = (int)((tensor_k & 0xFFFFu) << 16);           // tensor_dim0[15:0]
  g1[2] = (int)((tensor_k >> 16) | ((tensor_m & 0xFFFFu) << 16));
  g1[3] = (int)((tensor_m >> 16) | (tile_k << 16));    // tile_dim0
  g1[4] = (int)(tile_m & 0xFFFFu);                     // tile_dim1 (tile_dim2=0)
  g1[5] = (int)(stride_k & 0xFFFFFFFFu);               // tensor_dim0_stride lo
  g1[6] = (int)((stride_k >> 32) & 0xFFFFu);           // stride hi (dim1_stride=0)
  g1[7] = 0;
  tdm_i4 gz = {0, 0, 0, 0};
#if defined(__clang_major__) && (__clang_major__ >= 23)
  tdm_i8 gz8 = {0, 0, 0, 0, 0, 0, 0, 0};
  __builtin_amdgcn_tensor_load_to_lds(g0, g1, gz, gz, gz8, 0);
#else
  __builtin_amdgcn_tensor_load_to_lds(g0, g1, gz, gz, 0);
#endif
}
#endif  // HAVE_TDM

// ------------------------- fp32 -> bf16 pack -------------------------------
__global__ void cvt_f32_bf16(const float* __restrict__ src, u16* __restrict__ dst, int n) {
  int i = blockIdx.x * blockDim.x + threadIdx.x;
  if (i < n) dst[i] = f2bf(src[i]);
}

// ------------------ pipelined batched bf16 GEMM ----------------------------
// C[M,N] fp32 = scale * ( sum_b A_b[M,K] @ B_b[K,N]  +  sum_r bias[r][N] )
// Block tile 128x128, BK=32, 256 threads = 8 waves (2m x 4n);
// each wave owns 64x32 = 4x2 WMMA accumulators.
// Double-buffered LDS; A tiles via TDM, B tiles via global->reg->LDS(transposed).
__global__ void __launch_bounds__(256)
gemm_bf16(const u16* __restrict__ Abase, const u16* __restrict__ Bbase,
          const float* __restrict__ bias, int biasRows, float outScale,
          float* __restrict__ C, int M, int N, int K, int nBatch,
          size_t strideA, size_t strideB) {
  __shared__ __align__(16) u16 lA[2][128 * 32];     // [m][k] (TDM destination)
  __shared__ __align__(16) u16 lBt[2][128 * 32];    // transposed [n][k]
  const int tid  = threadIdx.x;
  const int wave = tid >> 5, lane = tid & 31;
  const int hl   = lane >> 4, l16 = lane & 15;
  const int wm   = wave & 1, wn = wave >> 1;
  const int mBase = blockIdx.y * 128;
  const int n0    = blockIdx.x * 128;
  const int ktiles = K >> 5;
  const int nTiles = nBatch * ktiles;
  const int bk = tid >> 3, bc = (tid & 7) * 16;     // B-staging coords

  v8f acc[4][2];
#pragma unroll
  for (int i = 0; i < 4; ++i)
#pragma unroll
    for (int j = 0; j < 2; ++j) acc[i][j] = v8f_zero();

#ifdef HAVE_TDM
  const unsigned ldsA0 = (unsigned)(size_t)(void*)lA[0];
  const unsigned ldsA1 = (unsigned)(size_t)(void*)lA[1];
#endif

  // ---- prologue: stage tile 0 into buffer 0 ----
  {
#ifdef HAVE_TDM
    if (wave == 0)
      tdm_load_2d_to_lds(ldsA0, Abase + (size_t)mBase * K,
                         32u, 128u, (unsigned)K, (unsigned)M,
                         (unsigned long long)K);
#else
    {
      int row = tid >> 1, c2 = (tid & 1) * 16;
      const u16* src = Abase + (size_t)(mBase + row) * K + c2;
      *(float4*)&lA[0][row * 32 + c2]     = *(const float4*)src;
      *(float4*)&lA[0][row * 32 + c2 + 8] = *(const float4*)(src + 8);
    }
#endif
    const u16* srcB = Bbase + (size_t)bk * N + n0 + bc;
    Frag16 tb;
    tb.f4[0] = *(const float4*)srcB;
    tb.f4[1] = *(const float4*)(srcB + 8);
#pragma unroll
    for (int j = 0; j < 16; ++j) lBt[0][(bc + j) * 32 + bk] = tb.u[j];
#ifdef HAVE_TDM
    if (wave == 0) __builtin_amdgcn_s_wait_tensorcnt(0);
#endif
    __syncthreads();
  }

  // ---- pipelined main loop: one barrier per tile ----
  for (int t = 0; t < nTiles; ++t) {
    const int cur = t & 1, nxt = cur ^ 1;
    const bool hasNext = (t + 1) < nTiles;
    Frag16 bNext;
#ifndef HAVE_TDM
    Frag16 aNext;
#endif
    if (hasNext) {
      const int t1 = t + 1;
      const int br1 = t1 / ktiles, k01 = (t1 - br1 * ktiles) << 5;
#ifdef HAVE_TDM
      if (wave == 0)   // DMA next A tile while this tile computes
        tdm_load_2d_to_lds(nxt ? ldsA1 : ldsA0,
                           Abase + (size_t)br1 * strideA + (size_t)mBase * K + k01,
                           32u, 128u, (unsigned)K, (unsigned)M,
                           (unsigned long long)K);
#else
      {
        int row = tid >> 1, c2 = (tid & 1) * 16;
        const u16* src = Abase + (size_t)br1 * strideA +
                         (size_t)(mBase + row) * K + k01 + c2;
        aNext.f4[0] = *(const float4*)src;
        aNext.f4[1] = *(const float4*)(src + 8);
      }
#endif
      const u16* srcB = Bbase + (size_t)br1 * strideB + (size_t)(k01 + bk) * N + n0 + bc;
      bNext.f4[0] = *(const float4*)srcB;
      bNext.f4[1] = *(const float4*)(srcB + 8);
      if (t + 2 < nTiles) {
        const int t2 = t + 2;
        const int br2 = t2 / ktiles, k02 = (t2 - br2 * ktiles) << 5;
        __builtin_prefetch(Bbase + (size_t)br2 * strideB + (size_t)(k02 + bk) * N + n0 + bc,
                           0, 1);
      }
    }

    // ---- compute on current buffers ----
    Frag16 af[4];
#pragma unroll
    for (int fm = 0; fm < 4; ++fm) {
      const int mloc = wm * 64 + fm * 16 + l16;
      af[fm].f4[0] = *(const float4*)&lA[cur][mloc * 32 + 16 * hl];
      af[fm].f4[1] = *(const float4*)&lA[cur][mloc * 32 + 16 * hl + 8];
    }
#pragma unroll
    for (int fn = 0; fn < 2; ++fn) {
      Frag16 bfr;
      const int nloc = wn * 32 + fn * 16 + l16;
      bfr.f4[0] = *(const float4*)&lBt[cur][nloc * 32 + 16 * hl];
      bfr.f4[1] = *(const float4*)&lBt[cur][nloc * 32 + 16 * hl + 8];
#pragma unroll
      for (int fm = 0; fm < 4; ++fm)
        acc[fm][fn] = __builtin_amdgcn_wmma_f32_16x16x32_bf16(
            false, af[fm].v, false, bfr.v, (short)0, acc[fm][fn], false, false);
    }

    // ---- publish next buffers, drain TDM, single barrier ----
    if (hasNext) {
#pragma unroll
      for (int j = 0; j < 16; ++j) lBt[nxt][(bc + j) * 32 + bk] = bNext.u[j];
#ifdef HAVE_TDM
      if (wave == 0) __builtin_amdgcn_s_wait_tensorcnt(0);
#else
      {
        int row = tid >> 1, c2 = (tid & 1) * 16;
        *(float4*)&lA[nxt][row * 32 + c2]     = aNext.f4[0];
        *(float4*)&lA[nxt][row * 32 + c2 + 8] = aNext.f4[1];
      }
#endif
    }
    __syncthreads();
  }

  // ---- epilogue: bias (sum of biasRows rows) and scale ----
#pragma unroll
  for (int fn = 0; fn < 2; ++fn) {
    int n = n0 + wn * 32 + fn * 16 + l16;
    float bsum = 0.f;
    for (int rb = 0; rb < biasRows; ++rb) bsum += bias[(size_t)rb * N + n];
#pragma unroll
    for (int fm = 0; fm < 4; ++fm)
#pragma unroll
      for (int r = 0; r < 8; ++r) {
        int m = mBase + wm * 64 + fm * 16 + r + 8 * hl;
        C[(size_t)m * N + n] = outScale * (acc[fm][fn][r] + bsum);
      }
  }
}

// --------------------- RoPE + repack to head-major bf16 --------------------
__global__ void rope_pack_q(const float* __restrict__ Qf, u16* __restrict__ Qbf) {
  int idx = blockIdx.x * blockDim.x + threadIdx.x;   // HTOT*T*32
  int i = idx & 31;
  int t = (idx >> 5) & (T_SEQ - 1);
  int h = idx >> 15;
  float invf = __powf(10000.f, -(2.f * (float)i) / (float)DHEAD);
  float ang = (float)t * invf;
  float c = __cosf(ang), s = __sinf(ang);
  const float* src = Qf + (size_t)t * QDIM + h * DHEAD;
  float x1 = src[2 * i], x2 = src[2 * i + 1];
  u16* dst = Qbf + ((size_t)h * T_SEQ + t) * DHEAD;
  dst[i]      = f2bf(x1 * c - x2 * s);
  dst[i + 32] = f2bf(x1 * s + x2 * c);
}

__global__ void rope_pack_k(const float* __restrict__ Kf, u16* __restrict__ Kbf) {
  int idx = blockIdx.x * blockDim.x + threadIdx.x;   // NHEAD*T*32
  int i = idx & 31;
  int t = (idx >> 5) & (T_SEQ - 1);
  int nh = idx >> 15;
  float invf = __powf(10000.f, -(2.f * (float)i) / (float)DHEAD);
  float ang = (float)t * invf;
  float c = __cosf(ang), s = __sinf(ang);
  const float* src = Kf + (size_t)t * DMODEL + nh * DHEAD;
  float x1 = src[2 * i], x2 = src[2 * i + 1];
  u16* dst = Kbf + ((size_t)nh * T_SEQ + t) * DHEAD;
  dst[i]      = f2bf(x1 * c - x2 * s);
  dst[i + 32] = f2bf(x1 * s + x2 * c);
}

// key_self = sum_d k^2 is RoPE-invariant; compute rsqrt(clip(.,1e-6)) pre-rope
__global__ void key_self_kernel(const float* __restrict__ Kf, float* __restrict__ invks) {
  int gw = (int)((blockIdx.x * blockDim.x + threadIdx.x) >> 5);  // one wave per (nh,t)
  int lane = threadIdx.x & 31;
  int nh = gw >> 10, t = gw & (T_SEQ - 1);
  const float* k = Kf + (size_t)t * DMODEL + nh * DHEAD;
  float a = k[lane], b = k[lane + 32];
  float s = a * a + b * b;
#pragma unroll
  for (int off = 16; off > 0; off >>= 1) s += __shfl_xor(s, off, 32);
  if (lane == 0) invks[nh * T_SEQ + t] = rsqrtf(fmaxf(s, 1e-6f));
}

// V repacked transposed: Vt[nh][d][t] so P@V B-fragments are contiguous b128
__global__ void pack_vt(const float* __restrict__ Vf, u16* __restrict__ Vt) {
  int idx = blockIdx.x * blockDim.x + threadIdx.x;   // NHEAD*DHEAD*T
  int t = idx & (T_SEQ - 1);
  int d = (idx >> 10) & (DHEAD - 1);
  int nh = idx >> 16;
  Vt[idx] = f2bf(Vf[(size_t)t * DMODEL + nh * DHEAD + d]);
}

// --------------------------- attention core --------------------------------
// grid: (T/64, HTOT), block 128 (4 waves). Wave handles 16 query rows.
__global__ void __launch_bounds__(128)
attn_kernel(const u16* __restrict__ Qbf, const u16* __restrict__ Kbf,
            const u16* __restrict__ Vt, const float* __restrict__ invks,
            const float* __restrict__ sinks, const float* __restrict__ vnull,
            float* __restrict__ ctx) {
  __shared__ __align__(16) u16 pStage[4][16 * 32];   // wave-private P tiles
  const int wave = threadIdx.x >> 5, lane = threadIdx.x & 31;
  const int hl = lane >> 4, l16 = lane & 15;
  const int h = blockIdx.y, nh = h & 15, br = h >> 4;
  const int q0 = blockIdx.x * 64 + wave * 16;
  const int qmax = q0 + 15;
  const float sink = sinks[h];
  const float attnscale = 0.125f;                 // DHEAD^-0.5
  const float sscale = 1.8137993642342178f;       // pi/sqrt(3)

  // persistent Q A-fragments (16 rows x 64 dh = 2 chunks of K=32)
  Frag16 aQ[2];
  {
    const u16* qp = Qbf + ((size_t)h * T_SEQ + (q0 + l16)) * DHEAD;
#pragma unroll
    for (int c = 0; c < 2; ++c) {
      aQ[c].f4[0] = *(const float4*)(qp + c * 32 + 16 * hl);
      aQ[c].f4[1] = *(const float4*)(qp + c * 32 + 16 * hl + 8);
    }
  }

  v8f acc[4] = {v8f_zero(), v8f_zero(), v8f_zero(), v8f_zero()};
  float rs[8] = {0.f, 0.f, 0.f, 0.f, 0.f, 0.f, 0.f, 0.f};
  u16* pw = pStage[wave];

  for (int k0 = 0; k0 <= qmax; k0 += 32) {
    if (k0 + 32 <= qmax)   // prefetch next K block (global_prefetch_b8)
      __builtin_prefetch(Kbf + ((size_t)nh * T_SEQ + (k0 + 32 + l16)) * DHEAD, 0, 1);
#pragma unroll
    for (int s = 0; s < 2; ++s) {
      const int kb = k0 + s * 16;
      const int n = kb + l16;          // key column for this lane
      if (kb <= qmax) {
        v8f sc = v8f_zero();
        const u16* kp = Kbf + ((size_t)nh * T_SEQ + n) * DHEAD;
#pragma unroll
        for (int c = 0; c < 2; ++c) {  // S = Q @ K^T over dh
          Frag16 bk;
          bk.f4[0] = *(const float4*)(kp + c * 32 + 16 * hl);
          bk.f4[1] = *(const float4*)(kp + c * 32 + 16 * hl + 8);
          sc = __builtin_amdgcn_wmma_f32_16x16x32_bf16(
              false, aQ[c].v, false, bk.v, (short)0, sc, false, false);
        }
        const float ik = invks[nh * T_SEQ + n] * attnscale;
#pragma unroll
        for (int r = 0; r < 8; ++r) {
          int m = q0 + r + 8 * hl;
          float w = 0.f;
          if (n <= m) {                            // causal
            float x = sc[r] * ik;
            float sp = (x > 20.f) ? x : log1pf(__expf(x));   // softplus
            w = sp * (1.f / (1.f + __expf(-sscale * sp)));   // silu gate
            if (w < sink) w = 0.f;                 // sink threshold
          }
          rs[r] += w;
          pw[(r + 8 * hl) * 32 + s * 16 + l16] = f2bf(w);
        }
      } else {
#pragma unroll
        for (int r = 0; r < 8; ++r)
          pw[(r + 8 * hl) * 32 + s * 16 + l16] = 0;
      }
    }
    // P(16x32) @ V(32x64); same-wave LDS ops are in-order -> no barrier needed
    Frag16 ap;
    ap.f4[0] = *(const float4*)&pw[l16 * 32 + 16 * hl];
    ap.f4[1] = *(const float4*)&pw[l16 * 32 + 16 * hl + 8];
#pragma unroll
    for (int ns = 0; ns < 4; ++ns) {
      int d = ns * 16 + l16;
      const u16* vp = Vt + ((size_t)nh * DHEAD + d) * T_SEQ + k0 + 16 * hl;
      Frag16 bv;
      bv.f4[0] = *(const float4*)vp;
      bv.f4[1] = *(const float4*)(vp + 8);
      acc[ns] = __builtin_amdgcn_wmma_f32_16x16x32_bf16(
          false, ap.v, false, bv.v, (short)0, acc[ns], false, false);
    }
  }

  // row sums: reduce across the 16 lanes of each half
#pragma unroll
  for (int r = 0; r < 8; ++r) {
    float v = rs[r];
    v += __shfl_xor(v, 1, 32);
    v += __shfl_xor(v, 2, 32);
    v += __shfl_xor(v, 4, 32);
    v += __shfl_xor(v, 8, 32);
    rs[r] = v + sink + 1e-6f;
  }
#pragma unroll
  for (int ns = 0; ns < 4; ++ns) {
    int d = ns * 16 + l16;
    float vn = vnull[h * DHEAD + d];   // (NBR,DMODEL) flat == (HTOT,DHEAD) flat
#pragma unroll
    for (int r = 0; r < 8; ++r) {
      int m = q0 + r + 8 * hl;
      float inv = 1.f / rs[r];
      ctx[((size_t)br * T_SEQ + m) * DMODEL + nh * DHEAD + d] =
          acc[ns][r] * inv + sink * inv * vn;
    }
  }
}

// ---------------------------------------------------------------------------
extern "C" void kernel_launch(void* const* d_in, const int* in_sizes, int n_in,
                              void* d_out, int out_size, void* d_ws, size_t ws_size,
                              hipStream_t stream) {
  (void)in_sizes; (void)n_in; (void)out_size; (void)ws_size;
  const float* X        = (const float*)d_in[0];
  const float* W_Q      = (const float*)d_in[1];
  const float* b_Q      = (const float*)d_in[2];
  const float* W_K      = (const float*)d_in[3];
  const float* b_K      = (const float*)d_in[4];
  const float* W_V      = (const float*)d_in[5];
  const float* b_V      = (const float*)d_in[6];
  const float* sinks    = (const float*)d_in[7];
  const float* v_nulls  = (const float*)d_in[8];
  const float* W_O      = (const float*)d_in[9];
  const float* W_O_bias = (const float*)d_in[10];
  float* out = (float*)d_out;

  char* ws = (char*)d_ws;
  size_t off = 0;
  auto alloc = [&](size_t bytes) -> void* {
    void* p = ws + off;
    off += (bytes + 255) & ~(size_t)255;
    return p;
  };

  u16*   Xbf   = (u16*)  alloc((size_t)T_SEQ * DMODEL * 2);
  u16*   WQw   = (u16*)  alloc((size_t)DMODEL * QDIM * 2);
  u16*   WKw   = (u16*)  alloc((size_t)DMODEL * DMODEL * 2);
  u16*   WVw   = (u16*)  alloc((size_t)DMODEL * DMODEL * 2);
  u16*   WOw   = (u16*)  alloc((size_t)NBR * DMODEL * DMODEL * 2);
  float* Qf    = (float*)alloc((size_t)T_SEQ * QDIM * 4);
  float* Kf    = (float*)alloc((size_t)T_SEQ * DMODEL * 4);
  float* Vf    = (float*)alloc((size_t)T_SEQ * DMODEL * 4);
  u16*   Qbf   = (u16*)  alloc((size_t)T_SEQ * QDIM * 2);
  u16*   Kbf   = (u16*)  alloc((size_t)NHEAD * T_SEQ * DHEAD * 2);
  u16*   Vtbf  = (u16*)  alloc((size_t)NHEAD * DHEAD * T_SEQ * 2);
  float* invks = (float*)alloc((size_t)NHEAD * T_SEQ * 4);
  float* ctxf  = (float*)alloc((size_t)NBR * T_SEQ * DMODEL * 4);
  u16*   ctxbf = (u16*)  alloc((size_t)NBR * T_SEQ * DMODEL * 2);

  auto cvt = [&](const float* s, u16* d, int n) {
    cvt_f32_bf16<<<(n + 255) / 256, 256, 0, stream>>>(s, d, n);
  };
  cvt(X,   Xbf, T_SEQ * DMODEL);
  cvt(W_Q, WQw, DMODEL * QDIM);
  cvt(W_K, WKw, DMODEL * DMODEL);
  cvt(W_V, WVw, DMODEL * DMODEL);
  cvt(W_O, WOw, NBR * DMODEL * DMODEL);

  dim3 blk256(256);
  // QKV projections: single batch, one bias row, scale 1
  gemm_bf16<<<dim3(QDIM / 128,   T_SEQ / 128), blk256, 0, stream>>>(
      Xbf, WQw, b_Q, 1, 1.0f, Qf, T_SEQ, QDIM,   DMODEL, 1, 0, 0);
  gemm_bf16<<<dim3(DMODEL / 128, T_SEQ / 128), blk256, 0, stream>>>(
      Xbf, WKw, b_K, 1, 1.0f, Kf, T_SEQ, DMODEL, DMODEL, 1, 0, 0);
  gemm_bf16<<<dim3(DMODEL / 128, T_SEQ / 128), blk256, 0, stream>>>(
      Xbf, WVw, b_V, 1, 1.0f, Vf, T_SEQ, DMODEL, DMODEL, 1, 0, 0);

  rope_pack_q<<<(HTOT  * T_SEQ * 32) / 256, 256, 0, stream>>>(Qf, Qbf);
  rope_pack_k<<<(NHEAD * T_SEQ * 32) / 256, 256, 0, stream>>>(Kf, Kbf);
  key_self_kernel<<<(NHEAD * T_SEQ * 32) / 256, 256, 0, stream>>>(Kf, invks);
  pack_vt<<<(NHEAD * DHEAD * T_SEQ) / 256, 256, 0, stream>>>(Vf, Vtbf);

  attn_kernel<<<dim3(T_SEQ / 64, HTOT), dim3(128), 0, stream>>>(Qbf, Kbf, Vtbf, invks,
                                                                sinks, v_nulls, ctxf);

  cvt(ctxf, ctxbf, NBR * T_SEQ * DMODEL);
  // Output projection: 4 branches summed, 4 bias rows summed, scale 0.25 (mean)
  gemm_bf16<<<dim3(DMODEL / 128, T_SEQ / 128), blk256, 0, stream>>>(
      ctxbf, WOw, W_O_bias, NBR, 0.25f, out, T_SEQ, DMODEL, DMODEL, NBR,
      (size_t)T_SEQ * DMODEL, (size_t)DMODEL * DMODEL);
}
